// GemNetTDecoder_66864050864371
// MI455X (gfx1250) — matrix operand order repaired and checked by reading, not compiled, not run on hardware
//
#include <hip/hip_runtime.h>
#include <hip/hip_bf16.h>

// GemNet-T decoder for MI455X (gfx1250), wave32 + WMMA bf16.
// All GEMMs (N=128) run on v_wmma_f32_16x16x32_bf16 with fp32 accumulation.
// HBM-bound (~2GB traffic vs ~200 GFLOP): bf16 resident activations, gathers
// and bias/relu/gate/residual/scatter fused into GEMM epilogues.
// Weights are pre-packed into per-lane WMMA B-fragment order so each lane
// fetches its fragment with one 32B vector load (no LDS B tile, no u16 merges).

typedef __attribute__((ext_vector_type(16))) __bf16 v16bf;
typedef __attribute__((ext_vector_type(8)))  __bf16 v8bf;
typedef __attribute__((ext_vector_type(8)))  float  v8f;

#define N_ATOMS_C 16384
#define EDGES_C   327680
#define LA_STRIDE 40   // elements; 80B rows, keeps both 16B fragment halves aligned

// ---- A fragment: 2x ds_load_b128 + shuffle (layout per CDNA5 ISA 7.12.2) ----
// lanes 0-15: K = 0..7 / 16..23 ; lanes 16-31: K = 8..15 / 24..31, row = lane&15.

__device__ __forceinline__ v16bf load_a_frag(const __bf16* lA, int lane, int rbase) {
  const int row = rbase + (lane & 15);
  const int off = (lane >> 4) ? 8 : 0;
  const v8bf lo = *(const v8bf*)(lA + row * LA_STRIDE + off);
  const v8bf hi = *(const v8bf*)(lA + row * LA_STRIDE + 16 + off);
  return __builtin_shufflevector(lo, hi, 0, 1, 2, 3, 4, 5, 6, 7,
                                 8, 9, 10, 11, 12, 13, 14, 15);
}

// ---- B fragment from packed weights: one 32B load per lane ------------------
// Wp[((kt*8 + nt)*32 + lane)*16 + j] = W[(kt*32 + (lane>=16?8:0) + (j<8?j:j+8))*128 + nt*16 + (lane&15)]

__device__ __forceinline__ v16bf load_b_frag(const __bf16* Wp, int kt, int nt, int lane) {
  return *(const v16bf*)(Wp + ((size_t)(kt * 8 + nt) * 32 + lane) * 16);
}

// pack W[K x 128] (f32) -> Wp (bf16, fragment order); one thread per element
__global__ void wpack_kernel(const float* __restrict__ W, __bf16* __restrict__ Wp, int K) {
  const int idx = blockIdx.x * blockDim.x + threadIdx.x;
  if (idx >= K * 128) return;
  const int j    = idx & 15;
  const int lane = (idx >> 4) & 31;
  const int t    = idx >> 9;            // kt*8 + nt
  const int nt   = t & 7, kt = t >> 3;
  const int n = nt * 16 + (lane & 15);
  const int k = kt * 32 + ((lane >> 4) ? 8 : 0) + (j < 8 ? j : j + 8);
  Wp[idx] = (__bf16)W[(size_t)k * 128 + n];
}

// ---- Generic tiled GEMM: out[M x 128] = epilogue(A[M x K] @ W[K x 128]) ----
// 256 threads = 8 waves; block does 32 rows x 128 cols; wave w -> cols [16w,16w+16),
// two 16x16 WMMA tiles per wave sharing one B fragment.

__global__ void __launch_bounds__(256)
gemm_k_128n(const __bf16* __restrict__ A, int K,
            const __bf16* __restrict__ Wp,
            const float*  __restrict__ bias, int relu,
            const float*  __restrict__ gate,
            const float*  __restrict__ residual,
            const int*    __restrict__ scatter_idx,
            float*        __restrict__ scatter_out,
            float*        __restrict__ outF,
            __bf16*       __restrict__ outB) {
  __shared__ __bf16 lA[32 * LA_STRIDE];
  const int row0 = blockIdx.x * 32;
  const int tid  = threadIdx.x;
  const int lane = tid & 31, wv = tid >> 5;
  const int sr = (tid * 4) >> 5, sc = (tid * 4) & 31;    // staging: 4 elems/thread
  const __bf16* arow = A + (size_t)(row0 + sr) * K + sc;
  v8f acc0 = {}, acc1 = {};
  const int nk = K >> 5;
  for (int kt = 0; kt < nk; ++kt) {
    *(uint2*)(&lA[sr * LA_STRIDE + sc]) = *(const uint2*)(arow + kt * 32);
    if (kt + 1 < nk) __builtin_prefetch(arow + kt * 32 + 32, 0, 1);
    __syncthreads();
    const v16bf bfv = load_b_frag(Wp, kt, wv, lane);
    const v16bf a0  = load_a_frag(lA, lane, 0);
    const v16bf a1  = load_a_frag(lA, lane, 16);
    acc0 = __builtin_amdgcn_wmma_f32_16x16x32_bf16(false, a0, false, bfv,
                                                   (short)0, acc0, false, false);
    acc1 = __builtin_amdgcn_wmma_f32_16x16x32_bf16(false, a1, false, bfv,
                                                   (short)0, acc1, false, false);
    __syncthreads();
  }
  // epilogue: C layout VGPR v -> row v (lanes 0-15) / v+8 (lanes 16-31), N=lane&15
  const int n     = wv * 16 + (lane & 15);
  const int mbase = (lane >> 4) ? 8 : 0;
  const v8f accs[2] = {acc0, acc1};
#pragma unroll
  for (int h = 0; h < 2; ++h) {
#pragma unroll
    for (int v = 0; v < 8; ++v) {
      const int grow = row0 + h * 16 + mbase + v;
      float val = accs[h][v];
      if (bias)     val += bias[n];
      if (relu)     val  = fmaxf(val, 0.f);
      if (gate)     val *= gate[(size_t)grow * 128 + n];
      if (residual) val += residual[(size_t)grow * 128 + n];
      if (scatter_idx) {
        atomicAdd(&scatter_out[(size_t)scatter_idx[grow] * 128 + n], val);
      } else {
        outF[(size_t)grow * 128 + n] = val;
        if (outB) outB[(size_t)grow * 128 + n] = (__bf16)val;
      }
    }
  }
}

// ---- Edge GEMM with fused concat gather: A = [h[src] | h[dst] | third] ------
// K = 384 fixed; out = (residual?) + relu(A @ W + bias). In-place safe for m:
// all reads of row e finish before the owning block's epilogue writes it.

__global__ void __launch_bounds__(256)
gemm_edge384(const __bf16* __restrict__ hB,
             const int*    __restrict__ src,
             const int*    __restrict__ dst,
             const __bf16* __restrict__ third,
             const __bf16* __restrict__ Wp,
             const float*  __restrict__ bias,
             const float*  __restrict__ residual,
             float*        __restrict__ outF,
             __bf16*       __restrict__ outB) {
  __shared__ __bf16 lA[32 * LA_STRIDE];
  const int row0 = blockIdx.x * 32;
  const int tid  = threadIdx.x;
  const int lane = tid & 31, wv = tid >> 5;
  const int sr = (tid * 4) >> 5, sc = (tid * 4) & 31;
  const int e  = row0 + sr;
  const __bf16* gsrc = hB + (size_t)src[e] * 128 + sc;   // kc in [0,128)
  const __bf16* gdst = hB + (size_t)dst[e] * 128 + sc;   // kc in [128,256)
  const __bf16* gthd = third + (size_t)e * 128 + sc;     // kc in [256,384)
  v8f acc0 = {}, acc1 = {};
  for (int kt = 0; kt < 12; ++kt) {
    const int kc = kt * 32;
    const __bf16* g;
    if (kc < 128)      g = gsrc + kc;
    else if (kc < 256) g = gdst + (kc - 128);
    else               g = gthd + (kc - 256);
    *(uint2*)(&lA[sr * LA_STRIDE + sc]) = *(const uint2*)g;
    __syncthreads();
    const v16bf bfv = load_b_frag(Wp, kt, wv, lane);
    const v16bf a0  = load_a_frag(lA, lane, 0);
    const v16bf a1  = load_a_frag(lA, lane, 16);
    acc0 = __builtin_amdgcn_wmma_f32_16x16x32_bf16(false, a0, false, bfv,
                                                   (short)0, acc0, false, false);
    acc1 = __builtin_amdgcn_wmma_f32_16x16x32_bf16(false, a1, false, bfv,
                                                   (short)0, acc1, false, false);
    __syncthreads();
  }
  const int n     = wv * 16 + (lane & 15);
  const int mbase = (lane >> 4) ? 8 : 0;
  const v8f accs[2] = {acc0, acc1};
#pragma unroll
  for (int h = 0; h < 2; ++h) {
#pragma unroll
    for (int v = 0; v < 8; ++v) {
      const int grow = row0 + h * 16 + mbase + v;
      float val = fmaxf(accs[h][v] + bias[n], 0.f);
      if (residual) val += residual[(size_t)grow * 128 + n];
      outF[(size_t)grow * 128 + n] = val;
      outB[(size_t)grow * 128 + n] = (__bf16)val;
    }
  }
}

// ---- Scalar helper kernels --------------------------------------------------

__global__ void cart_kernel(const float* __restrict__ frac,
                            const float* __restrict__ lengths,
                            const float* __restrict__ angles,
                            float* __restrict__ cart) {
  const int a = blockIdx.x * blockDim.x + threadIdx.x;
  if (a >= N_ATOMS_C) return;
  const int cidx = a >> 6;   // 64 atoms per crystal
  const float D2R = 0.017453292519943295f;
  const float r0 = angles[cidx * 3 + 0] * D2R;
  const float r1 = angles[cidx * 3 + 1] * D2R;
  const float r2 = angles[cidx * 3 + 2] * D2R;
  const float c0 = cosf(r0), c1 = cosf(r1), c2 = cosf(r2), sg = sinf(r2);
  const float La = lengths[cidx * 3 + 0], Lb = lengths[cidx * 3 + 1], Lc = lengths[cidx * 3 + 2];
  const float cx = Lc * c1;
  const float cy = Lc * (c0 - c1 * c2) / sg;
  const float cz = sqrtf(fmaxf(Lc * Lc - cx * cx - cy * cy, 1e-8f));
  const float L[3][3] = {{La, 0.f, 0.f}, {Lb * c2, Lb * sg, 0.f}, {cx, cy, cz}};
  const float f0 = frac[a * 3], f1 = frac[a * 3 + 1], f2 = frac[a * 3 + 2];
#pragma unroll
  for (int j = 0; j < 3; ++j)
    cart[a * 3 + j] = f0 * L[0][j] + f1 * L[1][j] + f2 * L[2][j];
}

__global__ void __launch_bounds__(128)
edge_rbf_kernel(const float* __restrict__ cart,
                const int* __restrict__ src, const int* __restrict__ dst,
                float* __restrict__ unit, __bf16* __restrict__ rbfB) {
  const int e = blockIdx.x;
  const int s = src[e], d = dst[e];
  float dv[3];
  dv[0] = cart[d * 3 + 0] - cart[s * 3 + 0];
  dv[1] = cart[d * 3 + 1] - cart[s * 3 + 1];
  dv[2] = cart[d * 3 + 2] - cart[s * 3 + 2];
  const float dist = sqrtf(dv[0] * dv[0] + dv[1] * dv[1] + dv[2] * dv[2] + 1e-12f);
  const int j = threadIdx.x;
  const float delta = 6.0f / 127.0f;
  const float mu = j * delta;
  const float env = 0.5f * (cosf(3.14159265358979f * fminf(dist / 6.0f, 1.0f)) + 1.0f);
  const float t = (dist - mu) / delta;
  rbfB[(size_t)e * 128 + j] = (__bf16)(expf(-t * t) * env);
  if (j < 3) unit[(size_t)e * 3 + j] = dv[j] / dist;
}

__global__ void atom_concat_kernel(const float* __restrict__ emb,
                                   const int* __restrict__ types,
                                   const float* __restrict__ z,
                                   __bf16* __restrict__ ainB) {
  const int idx = blockIdx.x * blockDim.x + threadIdx.x;
  if (idx >= N_ATOMS_C * 384) return;
  const int row = idx / 384, c = idx - row * 384;
  float v;
  if (c < 128) v = emb[types[row] * 128 + c];
  else         v = z[(row >> 6) * 256 + (c - 128)];
  ainB[idx] = (__bf16)v;
}

__global__ void f2bf_kernel(const float* __restrict__ in, __bf16* __restrict__ out, int n) {
  const int i = blockIdx.x * blockDim.x + threadIdx.x;
  if (i < n) out[i] = (__bf16)in[i];
}

__global__ void zero_kernel(float* __restrict__ p, int n) {
  const int i = blockIdx.x * blockDim.x + threadIdx.x;
  if (i < n) p[i] = 0.f;
}

// f = m @ W_force + b; out[dst] += f * unit  (one wave per edge, 8 edges/block)
__global__ void __launch_bounds__(256)
force_kernel(const float* __restrict__ mF, const float* __restrict__ Wf,
             const float* __restrict__ bf_, const int* __restrict__ dst,
             const float* __restrict__ unit, float* __restrict__ out) {
  const int wv = threadIdx.x >> 5, lane = threadIdx.x & 31;
  const size_t e = (size_t)blockIdx.x * 8 + wv;
  const float* row = mF + e * 128;
  float s = 0.f;
#pragma unroll
  for (int t = 0; t < 4; ++t) s += row[lane + 32 * t] * Wf[lane + 32 * t];
  s += __shfl_xor(s, 16);
  s += __shfl_xor(s, 8);
  s += __shfl_xor(s, 4);
  s += __shfl_xor(s, 2);
  s += __shfl_xor(s, 1);
  const float f = s + bf_[0];
  if (lane < 3) atomicAdd(&out[(size_t)dst[e] * 3 + lane], f * unit[e * 3 + lane]);
}

__global__ void fc3_kernel(const float* __restrict__ a2, const float* __restrict__ W,
                           const float* __restrict__ b, float* __restrict__ out) {
  const int idx = blockIdx.x * blockDim.x + threadIdx.x;
  if (idx >= N_ATOMS_C * 2) return;
  const int row = idx >> 1, c = idx & 1;
  float s = b[c];
  const float* ar = a2 + (size_t)row * 128;
  for (int k = 0; k < 128; ++k) s += ar[k] * W[k * 2 + c];
  out[idx] = s;
}

// ---- Host driver ------------------------------------------------------------

extern "C" void kernel_launch(void* const* d_in, const int* in_sizes, int n_in,
                              void* d_out, int out_size, void* d_ws, size_t ws_size,
                              hipStream_t stream) {
  (void)in_sizes; (void)n_in; (void)out_size; (void)ws_size;
  const float* z       = (const float*)d_in[0];
  const float* frac    = (const float*)d_in[1];
  const float* lengths = (const float*)d_in[2];
  const float* angles  = (const float*)d_in[3];
  const int*   types   = (const int*)d_in[4];
  const int*   src     = (const int*)d_in[5];
  const int*   dst     = (const int*)d_in[6];
  const float* emb     = (const float*)d_in[7];
  const float* W_in    = (const float*)d_in[8];
  const float* b_in    = (const float*)d_in[9];
  const float* W_edge  = (const float*)d_in[10];
  const float* b_edge  = (const float*)d_in[11];
  const float* Wb_rbf  = (const float*)d_in[12];
  const float* Wb_msg  = (const float*)d_in[13];
  const float* bb_msg  = (const float*)d_in[14];
  const float* Wb_atom = (const float*)d_in[15];
  const float* bb_atom = (const float*)d_in[16];
  const float* Wb_upd  = (const float*)d_in[17];
  const float* bb_upd  = (const float*)d_in[18];
  const float* W_force = (const float*)d_in[19];
  const float* b_force = (const float*)d_in[20];
  const float* W_fc1   = (const float*)d_in[21];
  const float* b_fc1   = (const float*)d_in[22];
  const float* W_fc2   = (const float*)d_in[23];
  const float* b_fc2   = (const float*)d_in[24];
  const float* W_fc3   = (const float*)d_in[25];
  const float* b_fc3   = (const float*)d_in[26];
  float* out = (float*)d_out;

  // workspace carve-out (256B aligned)
  char* base = (char*)d_ws;
  size_t off = 0;
  auto carve = [&](size_t bytes) -> char* {
    char* p = base + off;
    off = (off + bytes + 255) & ~(size_t)255;
    return p;
  };
  __bf16* winB   = (__bf16*)carve(49152 * 2);
  __bf16* wedgeB = (__bf16*)carve(49152 * 2);
  __bf16* wrbfB  = (__bf16*)carve(49152 * 2);     // 3 x (128x128)
  __bf16* wmsgB  = (__bf16*)carve(49152 * 2);
  __bf16* watomB = (__bf16*)carve(49152 * 2);
  __bf16* wupdB  = (__bf16*)carve(147456 * 2);    // 3 x (384x128)
  __bf16* wfc1B  = (__bf16*)carve(16384 * 2);
  __bf16* wfc2B  = (__bf16*)carve(16384 * 2);
  float*  cart   = (float*)carve((size_t)N_ATOMS_C * 3 * 4);
  float*  unit   = (float*)carve((size_t)EDGES_C * 3 * 4);
  __bf16* rbfB   = (__bf16*)carve((size_t)EDGES_C * 128 * 2);
  __bf16* ainB   = (__bf16*)carve((size_t)N_ATOMS_C * 384 * 2);
  float*  hF     = (float*)carve((size_t)N_ATOMS_C * 128 * 4);
  __bf16* hB     = (__bf16*)carve((size_t)N_ATOMS_C * 128 * 2);
  float*  mF     = (float*)carve((size_t)EDGES_C * 128 * 4);
  __bf16* mB     = (__bf16*)carve((size_t)EDGES_C * 128 * 2);
  float*  gateF  = (float*)carve((size_t)EDGES_C * 128 * 4);
  float*  aggF   = (float*)carve((size_t)N_ATOMS_C * 128 * 4);
  __bf16* aggB   = (__bf16*)carve((size_t)N_ATOMS_C * 128 * 2);
  float*  a1F    = (float*)carve((size_t)N_ATOMS_C * 128 * 4);
  __bf16* a1B    = (__bf16*)carve((size_t)N_ATOMS_C * 128 * 2);
  float*  a2F    = (float*)carve((size_t)N_ATOMS_C * 128 * 4);

  const int T = 256;
  auto cdiv = [](int a, int b) { return (a + b - 1) / b; };

  // 1) weights -> bf16, packed in WMMA B-fragment order
  wpack_kernel<<<cdiv(49152, T), T, 0, stream>>>(W_in,   winB,   384);
  wpack_kernel<<<cdiv(49152, T), T, 0, stream>>>(W_edge, wedgeB, 384);
  for (int i = 0; i < 3; ++i) {
    wpack_kernel<<<cdiv(16384, T), T, 0, stream>>>(Wb_rbf  + (size_t)i * 16384, wrbfB  + (size_t)i * 16384, 128);
    wpack_kernel<<<cdiv(16384, T), T, 0, stream>>>(Wb_msg  + (size_t)i * 16384, wmsgB  + (size_t)i * 16384, 128);
    wpack_kernel<<<cdiv(16384, T), T, 0, stream>>>(Wb_atom + (size_t)i * 16384, watomB + (size_t)i * 16384, 128);
    wpack_kernel<<<cdiv(49152, T), T, 0, stream>>>(Wb_upd  + (size_t)i * 49152, wupdB  + (size_t)i * 49152, 384);
  }
  wpack_kernel<<<cdiv(16384, T), T, 0, stream>>>(W_fc1, wfc1B, 128);
  wpack_kernel<<<cdiv(16384, T), T, 0, stream>>>(W_fc2, wfc2B, 128);

  // 2) geometry
  cart_kernel<<<cdiv(N_ATOMS_C, T), T, 0, stream>>>(frac, lengths, angles, cart);
  edge_rbf_kernel<<<EDGES_C, 128, 0, stream>>>(cart, src, dst, unit, rbfB);

  // 3) h = relu([emb | z] @ W_in + b_in)
  atom_concat_kernel<<<cdiv(N_ATOMS_C * 384, T), T, 0, stream>>>(emb, types, z, ainB);
  gemm_k_128n<<<N_ATOMS_C / 32, T, 0, stream>>>(ainB, 384, winB, b_in, 1,
      nullptr, nullptr, nullptr, nullptr, hF, hB);

  // 4) m = relu([h[src] | h[dst] | rbf] @ W_edge + b_edge)
  gemm_edge384<<<EDGES_C / 32, T, 0, stream>>>(hB, src, dst, rbfB, wedgeB,
      b_edge, nullptr, mF, mB);

  // 5) message-passing blocks
  for (int i = 0; i < 3; ++i) {
    // gate = rbf @ Wb_rbf[i]
    gemm_k_128n<<<EDGES_C / 32, T, 0, stream>>>(rbfB, 128, wrbfB + (size_t)i * 16384,
        nullptr, 0, nullptr, nullptr, nullptr, nullptr, gateF, nullptr);
    // agg = segment_sum(relu(m@Wmsg+b)*gate, dst)  -- fused scatter atomics
    zero_kernel<<<cdiv(N_ATOMS_C * 128, T), T, 0, stream>>>(aggF, N_ATOMS_C * 128);
    gemm_k_128n<<<EDGES_C / 32, T, 0, stream>>>(mB, 128, wmsgB + (size_t)i * 16384,
        bb_msg + i * 128, 1, gateF, nullptr, dst, aggF, nullptr, nullptr);
    // h += relu(agg @ Watom + b)
    f2bf_kernel<<<cdiv(N_ATOMS_C * 128, T), T, 0, stream>>>(aggF, aggB, N_ATOMS_C * 128);
    gemm_k_128n<<<N_ATOMS_C / 32, T, 0, stream>>>(aggB, 128, watomB + (size_t)i * 16384,
        bb_atom + i * 128, 1, nullptr, hF, nullptr, nullptr, hF, hB);
    // m += relu([h[src] | h[dst] | m] @ Wupd + b)   (in-place, block-local safe)
    gemm_edge384<<<EDGES_C / 32, T, 0, stream>>>(hB, src, dst, mB,
        wupdB + (size_t)i * 49152, bb_upd + i * 128, mF, mF, mB);
  }

  // 6) forces: out[:,0:3] scatter of (m@W_force+b)*unit
  zero_kernel<<<cdiv(N_ATOMS_C * 3, T), T, 0, stream>>>(out, N_ATOMS_C * 3);
  force_kernel<<<EDGES_C / 8, T, 0, stream>>>(mF, W_force, b_force, dst, unit, out);

  // 7) atom-type head
  gemm_k_128n<<<N_ATOMS_C / 32, T, 0, stream>>>(hB, 128, wfc1B, b_fc1, 1,
      nullptr, nullptr, nullptr, nullptr, a1F, a1B);
  gemm_k_128n<<<N_ATOMS_C / 32, T, 0, stream>>>(a1B, 128, wfc2B, b_fc2, 1,
      nullptr, nullptr, nullptr, nullptr, a2F, nullptr);
  fc3_kernel<<<cdiv(N_ATOMS_C * 2, T), T, 0, stream>>>(a2F, W_fc3, b_fc3,
      out + (size_t)N_ATOMS_C * 3);
}
